// DecoderLayer_24326694764883
// MI455X (gfx1250) — matrix operand order repaired and checked by reading, not compiled, not run on hardware
//
#include <hip/hip_runtime.h>

// ---------------------------------------------------------------------------
// Types for CDNA5 WMMA (wave32): D(16x16 f32) = A(16x32 bf16) x B(32x16 bf16) + C
// ---------------------------------------------------------------------------
typedef __attribute__((ext_vector_type(16))) __bf16 bf16x16;
typedef __attribute__((ext_vector_type(8)))  __bf16 bf16x8;
typedef __attribute__((ext_vector_type(8)))  float  f32x8;

__device__ __forceinline__ unsigned short bfbits(__bf16 b) {
    union { __bf16 b; unsigned short u; } c; c.b = b; return c.u;
}
__device__ __forceinline__ unsigned int pack2(float lo, float hi) {
    return (unsigned int)bfbits((__bf16)lo) | ((unsigned int)bfbits((__bf16)hi) << 16);
}
__device__ __forceinline__ bf16x16 cat16(bf16x8 lo, bf16x8 hi) {
    return __builtin_shufflevector(lo, hi, 0, 1, 2, 3, 4, 5, 6, 7,
                                   8, 9, 10, 11, 12, 13, 14, 15);
}
__device__ __forceinline__ bf16x8 lds_ld8(const unsigned short* p) {
    return *(const bf16x8*)p;     // 16B-aligned -> ds_load_b128
}
__device__ __forceinline__ bf16x8 g_ld8(const unsigned short* p) {
    return *(const bf16x8*)p;     // 16B-aligned -> global_load_b128
}
__device__ __forceinline__ f32x8 f8zero() {
    f32x8 z;
#pragma unroll
    for (int i = 0; i < 8; ++i) z[i] = 0.0f;
    return z;
}
__device__ __forceinline__ f32x8 wmma_bf16(bf16x16 a, bf16x16 b, f32x8 c) {
    return __builtin_amdgcn_wmma_f32_16x16x32_bf16(false, a, false, b,
                                                   (short)0, c, false, false);
}

// ---- gfx1250 async global->LDS copy (ASYNCcnt), ISA §15.18.3 op 98 ----------
__device__ __forceinline__ unsigned ldsOff(const void* p) {
    return (unsigned)(unsigned long long)p;   // low 32 bits of LDS aperture addr
}
__device__ __forceinline__ void async_copy16(unsigned lds_addr, const void* gaddr) {
    asm volatile("global_load_async_to_lds_b128 %0, %1, off"
                 :: "v"(lds_addr), "v"((unsigned long long)gaddr)
                 : "memory");
}
__device__ __forceinline__ void wait_async() {
    asm volatile("s_wait_asynccnt 0x0" ::: "memory");
}

// ---------------------------------------------------------------------------
// fp32 W[K,N] -> bf16 W^T[N,K]  (one-time; makes B fragments contiguous in K)
// ---------------------------------------------------------------------------
__global__ void cvtT_kernel(const float* __restrict__ s,
                            unsigned short* __restrict__ d, int Kdim, int Ndim) {
    int i = blockIdx.x * blockDim.x + threadIdx.x;      // i = k*N + n (coalesced read)
    if (i < Kdim * Ndim) {
        int k = i / Ndim, n = i % Ndim;
        d[(size_t)n * Kdim + k] = bfbits((__bf16)s[i]);
    }
}

// ---------------------------------------------------------------------------
// Tiled WMMA GEMM: C[M,N] = act(A[M,K]) @ W^T_bf16[N,K] + bias (+resid)(+relu)
// Block: 256 threads = 8 waves. Tile 128x128, K-step 32.
// LDS: As[m][k], Bs[n][k] (both 128x32 bf16, fragment-contiguous).
// B tile (and A tile when bf16) staged via global_load_async_to_lds_b128.
// ---------------------------------------------------------------------------
template <bool ABF>
__global__ __launch_bounds__(256) void gemm_kernel(
    const void* __restrict__ Ap, const unsigned short* __restrict__ Wt,
    const float* __restrict__ bias, const float* __restrict__ resid,
    float* __restrict__ outF, unsigned short* __restrict__ outB,
    int M, int N, int K, int relu) {
    __shared__ __align__(16) unsigned short As[128 * 32];
    __shared__ __align__(16) unsigned short Bs[128 * 32];

    const int tid  = threadIdx.x;
    const int wid  = tid >> 5;
    const int lane = tid & 31;
    const int l15  = lane & 15;
    const int hf   = (lane >> 4) & 1;

    const int row0 = blockIdx.y * 128;
    const int col0 = blockIdx.x * 128;
    const int wrow = (wid & 3) * 32;
    const int wcol = (wid >> 2) * 64;

    const float*          Af = (const float*)Ap;
    const unsigned short* Ab = (const unsigned short*)Ap;

    // staging coords: each thread owns 16 consecutive K elems of one row
    const int e0 = tid * 16;
    const int ar = e0 >> 5, ac = e0 & 31;      // row, k-offset (0/16)
    const unsigned ldsA = ldsOff(As) + tid * 32;
    const unsigned ldsB = ldsOff(Bs) + tid * 32;

    f32x8 acc[2][4];
#pragma unroll
    for (int i = 0; i < 2; ++i)
#pragma unroll
        for (int j = 0; j < 4; ++j) acc[i][j] = f8zero();

    for (int k0 = 0; k0 < K; k0 += 32) {
        // ---- B tile (weights, bf16): async copy straight into LDS
        {
            const unsigned short* src = Wt + (size_t)(col0 + ar) * K + k0 + ac;
            async_copy16(ldsB,      src);
            async_copy16(ldsB + 16, src + 8);
        }
        // ---- A tile
        if (ABF) {
            const unsigned short* src = Ab + (size_t)(row0 + ar) * K + k0 + ac;
            async_copy16(ldsA,      src);
            async_copy16(ldsA + 16, src + 8);
        } else {
            const float4* src = (const float4*)(Af + (size_t)(row0 + ar) * K + k0 + ac);
            float4 f0 = src[0], f1 = src[1], f2 = src[2], f3 = src[3];
            uint4 o0, o1;
            o0.x = pack2(f0.x, f0.y); o0.y = pack2(f0.z, f0.w);
            o0.z = pack2(f1.x, f1.y); o0.w = pack2(f1.z, f1.w);
            o1.x = pack2(f2.x, f2.y); o1.y = pack2(f2.z, f2.w);
            o1.z = pack2(f3.x, f3.y); o1.w = pack2(f3.z, f3.w);
            ((uint4*)As)[tid * 2]     = o0;
            ((uint4*)As)[tid * 2 + 1] = o1;
            if (k0 + 32 < K) __builtin_prefetch((const char*)src + 128, 0, 0);
        }
        wait_async();
        __syncthreads();

        // A fragments: per-lane two contiguous 8-elem chunks at k = ka, ka+16
        const int ka = hf * 8;
        bf16x16 af[2];
#pragma unroll
        for (int rt = 0; rt < 2; ++rt) {
            const int be = (wrow + rt * 16 + l15) * 32 + ka;
            af[rt] = cat16(lds_ld8(&As[be]), lds_ld8(&As[be + 16]));
        }
        // B fragments: two contiguous 8-elem chunks at k = kbb, kbb+16
        const int kbb = hf * 8;
#pragma unroll
        for (int ct = 0; ct < 4; ++ct) {
            const int be = (wcol + ct * 16 + l15) * 32 + kbb;
            bf16x16 bfr = cat16(lds_ld8(&Bs[be]), lds_ld8(&Bs[be + 16]));
#pragma unroll
            for (int rt = 0; rt < 2; ++rt)
                acc[rt][ct] = wmma_bf16(af[rt], bfr, acc[rt][ct]);
        }
        __syncthreads();
    }

    // epilogue: C layout row = vgpr + 8*half, col = lane&15
#pragma unroll
    for (int rt = 0; rt < 2; ++rt)
#pragma unroll
        for (int ct = 0; ct < 4; ++ct) {
            const int colc = col0 + wcol + ct * 16 + l15;
            const float bsv = bias ? bias[colc] : 0.0f;
#pragma unroll
            for (int r = 0; r < 8; ++r) {
                const int rowc = row0 + wrow + rt * 16 + r + hf * 8;
                float v = acc[rt][ct][r] + bsv;
                if (resid) v += resid[(size_t)rowc * N + colc];
                if (relu)  v = fmaxf(v, 0.0f);
                if (outB) outB[(size_t)rowc * N + colc] = bfbits((__bf16)v);
                else      outF[(size_t)rowc * N + colc] = v;
            }
        }
}

// ---------------------------------------------------------------------------
// Fused flash attention, all-bf16 operands. Q/K/V/O: [B*S, 512] bf16,
// head h at col h*64. Block = 128 threads = 4 waves; wave owns 16 query rows.
// K chunk (32x64) staged via async-to-LDS; V staged transposed.
// ---------------------------------------------------------------------------
__global__ __launch_bounds__(128) void attn_kernel(
    const unsigned short* __restrict__ Q, const unsigned short* __restrict__ K,
    const unsigned short* __restrict__ V, unsigned short* __restrict__ O,
    int S, int causal) {
    __shared__ __align__(16) unsigned short Ks[32 * 64];      // [key][d]
    __shared__ __align__(16) unsigned short Vt[64 * 32];      // [d][key]
    __shared__ __align__(16) unsigned short Ps[4][16 * 32];   // per-wave P [m][k]

    const int tid  = threadIdx.x;
    const int w    = tid >> 5;
    const int lane = tid & 31;
    const int l15  = lane & 15;
    const int hf   = (lane >> 4) & 1;

    const int h  = blockIdx.y;
    const int b  = blockIdx.z;
    const int q0 = blockIdx.x * 64 + w * 16;
    const size_t base = ((size_t)b * S) * 512 + (size_t)h * 64;

    // Q fragments (two 16x32 A tiles over DK=64) -- direct b128 global loads
    bf16x16 qa[2];
    const int ka = hf * 8;
#pragma unroll
    for (int dc = 0; dc < 2; ++dc) {
        const unsigned short* qp = Q + base + (size_t)(q0 + l15) * 512 + dc * 32 + ka;
        qa[dc] = cat16(g_ld8(qp), g_ld8(qp + 16));
    }

    f32x8 o[4];
#pragma unroll
    for (int t = 0; t < 4; ++t) o[t] = f8zero();
    float m_run[8], l_run[8];
#pragma unroll
    for (int r = 0; r < 8; ++r) { m_run[r] = -3.0e38f; l_run[r] = 0.0f; }

    const int kb = hf * 8;
    // staging coords: thread owns 16 consecutive d of one key row
    const int e0 = tid * 16;
    const int srow = e0 >> 6, scol = e0 & 63;      // key row, d offset
    const unsigned ldsK = ldsOff(Ks) + tid * 32;

    const int kend = causal ? (blockIdx.x * 64 + 64) : S;
    for (int kc = 0; kc < kend; kc += 32) {
        // ---- K chunk: async straight to LDS (already bf16)
        {
            const unsigned short* kg = K + base + (size_t)(kc + srow) * 512 + scol;
            async_copy16(ldsK,      kg);
            async_copy16(ldsK + 16, kg + 8);
        }
        // ---- V chunk: load + transpose-scatter into LDS
        {
            const uint4* vg = (const uint4*)(V + base + (size_t)(kc + srow) * 512 + scol);
            union { uint4 q; unsigned short s[8]; } v0, v1;
            v0.q = vg[0]; v1.q = vg[1];
#pragma unroll
            for (int i = 0; i < 8; ++i) {
                Vt[(scol + i) * 32 + srow]     = v0.s[i];
                Vt[(scol + 8 + i) * 32 + srow] = v1.s[i];
            }
        }
        wait_async();
        __syncthreads();

        // ---- scores: two 16x16 C tiles (keys kc..+15 and kc+16..+31)
        f32x8 s0 = f8zero(), s1 = f8zero();
#pragma unroll
        for (int dc = 0; dc < 2; ++dc) {
            const int be0 = l15 * 64 + dc * 32 + kb;
            const int be1 = (16 + l15) * 64 + dc * 32 + kb;
            bf16x16 bk0 = cat16(lds_ld8(&Ks[be0]), lds_ld8(&Ks[be0 + 16]));
            bf16x16 bk1 = cat16(lds_ld8(&Ks[be1]), lds_ld8(&Ks[be1 + 16]));
            s0 = wmma_bf16(qa[dc], bk0, s0);
            s1 = wmma_bf16(qa[dc], bk1, s1);
        }

        // ---- online softmax update
#pragma unroll
        for (int r = 0; r < 8; ++r) {
            float a0 = s0[r] * 0.125f;      // 1/sqrt(64)
            float a1 = s1[r] * 0.125f;
            const int rowq = q0 + r + hf * 8;
            if (causal) {
                if (kc + l15 > rowq)      a0 += -1.0e10f;
                if (kc + 16 + l15 > rowq) a1 += -1.0e10f;
            }
            float mx = fmaxf(a0, a1);
#pragma unroll
            for (int msk = 1; msk <= 8; msk <<= 1)
                mx = fmaxf(mx, __shfl_xor(mx, msk, 32));
            const float mn   = fmaxf(m_run[r], mx);
            const float corr = __expf(m_run[r] - mn);
            m_run[r] = mn;
            const float p0 = __expf(a0 - mn);
            const float p1 = __expf(a1 - mn);
            float rs = p0 + p1;
#pragma unroll
            for (int msk = 1; msk <= 8; msk <<= 1)
                rs += __shfl_xor(rs, msk, 32);
            l_run[r] = l_run[r] * corr + rs;
#pragma unroll
            for (int t = 0; t < 4; ++t) o[t][r] *= corr;
            Ps[w][(r + hf * 8) * 32 + l15]      = bfbits((__bf16)p0);
            Ps[w][(r + hf * 8) * 32 + 16 + l15] = bfbits((__bf16)p1);
        }
        __syncthreads();

        // ---- P fragment (contiguous chunks at k=ka, ka+16)
        const int pe = l15 * 32 + ka;
        bf16x16 pa = cat16(lds_ld8(&Ps[w][pe]), lds_ld8(&Ps[w][pe + 16]));

        // ---- O += P @ V (V fragments contiguous in Vt)
#pragma unroll
        for (int ct = 0; ct < 4; ++ct) {
            const int be = (ct * 16 + l15) * 32 + kb;
            bf16x16 bv = cat16(lds_ld8(&Vt[be]), lds_ld8(&Vt[be + 16]));
            o[ct] = wmma_bf16(pa, bv, o[ct]);
        }
        __syncthreads();
    }

    // ---- normalize and store (bf16)
#pragma unroll
    for (int r = 0; r < 8; ++r) {
        const float inv = 1.0f / l_run[r];
        const size_t orow = base + (size_t)(q0 + r + hf * 8) * 512;
#pragma unroll
        for (int ct = 0; ct < 4; ++ct)
            O[orow + ct * 16 + l15] = bfbits((__bf16)(o[ct][r] * inv));
    }
}

// ---------------------------------------------------------------------------
// [A,B,D] -> [B,A,D] transpose of rows (D=512, float4 vectorized)
// ---------------------------------------------------------------------------
__global__ __launch_bounds__(128) void transpose_kernel(
    const float4* __restrict__ in, float4* __restrict__ out, int Adim, int Bdim) {
    const int rid = blockIdx.x;
    const int a = rid / Bdim, b = rid % Bdim;
    const int d = threadIdx.x;            // 0..127 (512/4)
    out[(size_t)(b * Adim + a) * 128 + d] = in[(size_t)rid * 128 + d];
}

// ---------------------------------------------------------------------------
// LayerNorm over D=512: one wave per row, 8 rows per 256-thread block
// ---------------------------------------------------------------------------
__global__ __launch_bounds__(256) void ln_kernel(
    const float* __restrict__ X, const float* __restrict__ g,
    const float* __restrict__ bt, float* __restrict__ Y) {
    const int w = threadIdx.x >> 5, lane = threadIdx.x & 31;
    const size_t row = (size_t)blockIdx.x * 8 + w;
    const float* xr = X + row * 512;
    float v[16], s = 0.0f, ss = 0.0f;
#pragma unroll
    for (int i = 0; i < 4; ++i) {
        float4 f = *(const float4*)(xr + lane * 4 + i * 128);
        v[i * 4 + 0] = f.x; v[i * 4 + 1] = f.y; v[i * 4 + 2] = f.z; v[i * 4 + 3] = f.w;
        s  += f.x + f.y + f.z + f.w;
        ss += f.x * f.x + f.y * f.y + f.z * f.z + f.w * f.w;
    }
#pragma unroll
    for (int msk = 1; msk <= 16; msk <<= 1) {
        s  += __shfl_xor(s,  msk, 32);
        ss += __shfl_xor(ss, msk, 32);
    }
    const float mean = s * (1.0f / 512.0f);
    const float var  = ss * (1.0f / 512.0f) - mean * mean;
    const float rs   = rsqrtf(var + 1e-5f);
#pragma unroll
    for (int i = 0; i < 4; ++i) {
        const int c = lane * 4 + i * 128;
        float4 gg = *(const float4*)(g + c);
        float4 bb = *(const float4*)(bt + c);
        float4 y;
        y.x = (v[i * 4 + 0] - mean) * rs * gg.x + bb.x;
        y.y = (v[i * 4 + 1] - mean) * rs * gg.y + bb.y;
        y.z = (v[i * 4 + 2] - mean) * rs * gg.z + bb.z;
        y.w = (v[i * 4 + 3] - mean) * rs * gg.w + bb.w;
        *(float4*)(Y + row * 512 + c) = y;
    }
}

// ---------------------------------------------------------------------------
// host side
// ---------------------------------------------------------------------------
static void launch_gemm(bool abf, const void* A, const unsigned short* Wt,
                        const float* bias, const float* resid, float* oF,
                        unsigned short* oB, int M, int N, int K, int relu,
                        hipStream_t st) {
    dim3 g(N / 128, M / 128);
    if (abf) gemm_kernel<true ><<<g, 256, 0, st>>>(A, Wt, bias, resid, oF, oB, M, N, K, relu);
    else     gemm_kernel<false><<<g, 256, 0, st>>>(A, Wt, bias, resid, oF, oB, M, N, K, relu);
}

extern "C" void kernel_launch(void* const* d_in, const int* in_sizes, int n_in,
                              void* d_out, int out_size, void* d_ws, size_t ws_size,
                              hipStream_t stream) {
    (void)in_sizes; (void)n_in; (void)out_size; (void)ws_size;
    constexpr int T = 128, N = 256, D = 512, M = T * N, DFF = 2048;

    const float* x = (const float*)d_in[0];
    const float* sq_b = (const float*)d_in[2];
    const float* sk_b = (const float*)d_in[4];
    const float* sv_b = (const float*)d_in[6];
    const float* so_b = (const float*)d_in[8];
    const float* tq_b = (const float*)d_in[10];
    const float* tk_b = (const float*)d_in[12];
    const float* tv_b = (const float*)d_in[14];
    const float* to_b = (const float*)d_in[16];
    const float* ln_g = (const float*)d_in[17];
    const float* ln_b = (const float*)d_in[18];
    const float* f1_b = (const float*)d_in[20];
    const float* f2_b = (const float*)d_in[22];

    // --- workspace layout: bf16 W^T weights (8 MiB) + 5 x 64 MiB buffers
    char* wsb = (char*)d_ws;
    const int wsrc[10] = {1, 3, 5, 7, 9, 11, 13, 15, 19, 21};
    const int wK[10]   = {512, 512, 512, 512, 512, 512, 512, 512, 512, 2048};
    const int wN[10]   = {512, 512, 512, 512, 512, 512, 512, 512, 2048, 512};
    unsigned short* wb[10];
    size_t off = 0;
    for (int i = 0; i < 10; ++i) {
        wb[i] = (unsigned short*)(wsb + off);
        off += (size_t)wK[i] * wN[i] * 2;
    }
    const size_t WOFF = 8u << 20;              // 8 MiB
    const size_t SZB  = (size_t)M * D * 4;     // 64 MiB per buffer
    float* bufA = (float*)(wsb + WOFF);
    float* bufB = (float*)(wsb + WOFF + 1 * SZB);
    float* bufC = (float*)(wsb + WOFF + 2 * SZB);
    float* bufD = (float*)(wsb + WOFF + 3 * SZB);
    float* bufE = (float*)(wsb + WOFF + 4 * SZB);
    unsigned short* FF = (unsigned short*)bufD; // 128 MiB, spans D..E (dead then)

    // --- weight conversion (fp32 [K,N] -> bf16 [N,K])
    for (int i = 0; i < 10; ++i) {
        const int n = wK[i] * wN[i];
        cvtT_kernel<<<dim3((n + 255) / 256), 256, 0, stream>>>(
            (const float*)d_in[wsrc[i]], wb[i], wK[i], wN[i]);
    }

    // --- spatial attention (batch = T, seq = N, no mask); QKV/attn in bf16
    unsigned short* Qs = (unsigned short*)bufA;
    unsigned short* Ks = (unsigned short*)bufB;
    unsigned short* Vs = (unsigned short*)bufC;
    unsigned short* Os = (unsigned short*)bufD;
    launch_gemm(false, x, wb[0], sq_b, nullptr, nullptr, Qs, M, D, D, 0, stream);
    launch_gemm(false, x, wb[1], sk_b, nullptr, nullptr, Ks, M, D, D, 0, stream);
    launch_gemm(false, x, wb[2], sv_b, nullptr, nullptr, Vs, M, D, D, 0, stream);
    attn_kernel<<<dim3(N / 64, 8, T), 128, 0, stream>>>(Qs, Ks, Vs, Os, N, 0);
    launch_gemm(true, Os, wb[3], so_b, x, bufE, nullptr, M, D, D, 0, stream);      // x1

    // --- temporal attention (batch = N, seq = T, causal)
    transpose_kernel<<<dim3(M), 128, 0, stream>>>((const float4*)bufE,
                                                  (float4*)bufA, T, N);            // x1t
    unsigned short* Qt = (unsigned short*)bufB;
    unsigned short* Kt = (unsigned short*)bufC;
    unsigned short* Vt = (unsigned short*)bufD;
    unsigned short* Ot = (unsigned short*)bufE;
    launch_gemm(false, bufA, wb[4], tq_b, nullptr, nullptr, Qt, M, D, D, 0, stream);
    launch_gemm(false, bufA, wb[5], tk_b, nullptr, nullptr, Kt, M, D, D, 0, stream);
    launch_gemm(false, bufA, wb[6], tv_b, nullptr, nullptr, Vt, M, D, D, 0, stream);
    attn_kernel<<<dim3(T / 64, 8, N), 128, 0, stream>>>(Qt, Kt, Vt, Ot, T, 1);
    launch_gemm(true, Ot, wb[7], to_b, bufA, bufB, nullptr, M, D, D, 0, stream);   // x2t
    transpose_kernel<<<dim3(M), 128, 0, stream>>>((const float4*)bufB,
                                                  (float4*)bufC, N, T);            // x2

    // --- FFN with pre-norm
    ln_kernel<<<dim3(M / 8), 256, 0, stream>>>(bufC, ln_g, ln_b, bufA);            // h
    launch_gemm(false, bufA, wb[8], f1_b, nullptr, nullptr, FF, M, DFF, D, 1, stream);
    launch_gemm(true,  FF,   wb[9], f2_b, bufC, (float*)d_out, nullptr, M, D, DFF, 0, stream);
}
